// Block_35055523070295
// MI455X (gfx1250) — matrix operand order repaired and checked by reading, not compiled, not run on hardware
//
#include <hip/hip_runtime.h>

// ---------------------------------------------------------------------------
// Transformer block (pre-LN attention + MLP) for MI455X / gfx1250.
// All GEMMs via v_wmma_f32_16x16x32_bf16 (wave32). Flash-style attention,
// no score materialization. fp32 in/out, bf16 WMMA operands, fp32 accum.
// Register-blocked GEMMs: 32x64 per wave (8 WMMA per K-step, 6 operand loads).
// ---------------------------------------------------------------------------

typedef __bf16  v16bf __attribute__((ext_vector_type(16)));
typedef __bf16  v8bf  __attribute__((ext_vector_type(8)));
typedef float   v8f   __attribute__((ext_vector_type(8)));

#define BB 2
#define TT 2048
#define CC 768
#define HH 12
#define DD 64
#define MM (BB * TT)     // 4096 rows
#define FF (4 * CC)      // 3072

__device__ __forceinline__ v8f wmma_bf16(v16bf a, v16bf b, v8f c) {
  return __builtin_amdgcn_wmma_f32_16x16x32_bf16(false, a, false, b,
                                                 (short)0, c, false, false);
}

// A operand (16 rows x 32 K, bf16): lane = m + 16*khalf; 16 contiguous
// k values at k = 16*khalf.  p points at (row0, k0).
__device__ __forceinline__ v16bf load_a(const __bf16* p, int ld) {
  int lane = threadIdx.x & 31;
  const __bf16* r0 = p + (size_t)(lane & 15) * ld + ((lane >> 4) << 4);
  v8bf lo = *(const v8bf*)(r0);
  v8bf hi = *(const v8bf*)(r0 + 8);
  v16bf r;
#pragma unroll
  for (int i = 0; i < 8; ++i) { r[i] = lo[i]; r[i + 8] = hi[i]; }
  return r;
}

// B operand (32 K x 16 cols, bf16), loaded from N-major (transposed) storage:
// lane = n + 16*khalf; 16 contiguous k values at k = 16*khalf.
__device__ __forceinline__ v16bf load_b(const __bf16* p, int ld) {
  int lane = threadIdx.x & 31;
  const __bf16* r0 = p + (size_t)(lane & 15) * ld + ((lane >> 4) << 4);
  v8bf lo = *(const v8bf*)(r0);
  v8bf hi = *(const v8bf*)(r0 + 8);
  v16bf r;
#pragma unroll
  for (int i = 0; i < 8; ++i) { r[i] = lo[i]; r[i + 8] = hi[i]; }
  return r;
}

// ---------------------------------------------------------------------------
// Weight staging: fp32 (R x Cc) -> bf16 transposed (Cc x R), batched.
// ---------------------------------------------------------------------------
__global__ void transpose_cvt_kernel(const float* __restrict__ in,
                                     __bf16* __restrict__ out, int R, int Cc) {
  size_t base = (size_t)blockIdx.y * R * Cc;
  int idx = blockIdx.x * blockDim.x + threadIdx.x;
  if (idx < R * Cc) {
    int r = idx / Cc, c = idx % Cc;
    out[base + (size_t)c * R + r] = (__bf16)in[base + idx];
  }
}

// ---------------------------------------------------------------------------
// LayerNorm: fp32 (rows x C) -> bf16 (rows x C).  One block (256 thr) per row.
// ---------------------------------------------------------------------------
__global__ void ln_kernel(const float* __restrict__ x,
                          const float* __restrict__ g,
                          const float* __restrict__ b,
                          __bf16* __restrict__ out) {
  int row = blockIdx.x;
  const float* xr = x + (size_t)row * CC;
  float s = 0.f, ss = 0.f;
  for (int c = threadIdx.x; c < CC; c += 256) {
    float v = xr[c];
    s += v; ss += v * v;
  }
  __shared__ float red[512];
  red[threadIdx.x] = s;
  red[threadIdx.x + 256] = ss;
  __syncthreads();
  for (int off = 128; off > 0; off >>= 1) {
    if (threadIdx.x < off) {
      red[threadIdx.x]       += red[threadIdx.x + off];
      red[threadIdx.x + 256] += red[threadIdx.x + 256 + off];
    }
    __syncthreads();
  }
  float mu  = red[0] * (1.0f / CC);
  float var = red[256] * (1.0f / CC) - mu * mu;
  float rstd = rsqrtf(var + 1e-5f);
  for (int c = threadIdx.x; c < CC; c += 256)
    out[(size_t)row * CC + c] = (__bf16)((xr[c] - mu) * rstd * g[c] + b[c]);
}

// ---------------------------------------------------------------------------
// Fused QKV GEMM.  One wave per (bh, 32-row t-tile, dtile): computes a
// 32x16 tile of q, k and v (6 WMMA per K-step from 5 operand loads).
// h: (B*T, C) bf16.  wT: (H, D, C) bf16 (N-major).
// q,k: (B,H,T,D) bf16 row-major.  vT: (B,H,D,T) bf16 (pre-transposed for PV).
// ---------------------------------------------------------------------------
__global__ void qkv_kernel(const __bf16* __restrict__ h,
                           const __bf16* __restrict__ wqT,
                           const __bf16* __restrict__ wkT,
                           const __bf16* __restrict__ wvT,
                           __bf16* __restrict__ q, __bf16* __restrict__ k,
                           __bf16* __restrict__ vT) {
  int dtile = threadIdx.x >> 5;          // 4 waves = 4 d-tiles
  int lane  = threadIdx.x & 31;
  int ttile = blockIdx.x;                // 0..T/32-1
  int bh    = blockIdx.y;                // 0..B*H-1
  int b = bh / HH, hh = bh % HH;
  int tbase = ttile * 32;

  const __bf16* A  = h + ((size_t)(b * TT + tbase)) * CC;
  size_t wrow = (size_t)(hh * DD + dtile * 16) * CC;
  const __bf16* Bq = wqT + wrow;
  const __bf16* Bk = wkT + wrow;
  const __bf16* Bv = wvT + wrow;

  v8f aq[2] = {}, ak[2] = {}, av[2] = {};
  for (int c0 = 0; c0 < CC; c0 += 32) {
    v16bf a0 = load_a(A + c0, CC);
    v16bf a1 = load_a(A + (size_t)16 * CC + c0, CC);
    v16bf bq = load_b(Bq + c0, CC);
    v16bf bk = load_b(Bk + c0, CC);
    v16bf bv = load_b(Bv + c0, CC);
    aq[0] = wmma_bf16(a0, bq, aq[0]);
    aq[1] = wmma_bf16(a1, bq, aq[1]);
    ak[0] = wmma_bf16(a0, bk, ak[0]);
    ak[1] = wmma_bf16(a1, bk, ak[1]);
    av[0] = wmma_bf16(a0, bv, av[0]);
    av[1] = wmma_bf16(a1, bv, av[1]);
  }

  int mloc = (lane >> 4) << 3;           // 0 or 8
  int n = lane & 15;
  int d = dtile * 16 + n;
  size_t qbase = (size_t)bh * TT;
#pragma unroll
  for (int i = 0; i < 2; ++i) {
#pragma unroll
    for (int r = 0; r < 8; ++r) {
      int t = tbase + i * 16 + mloc + r;
      q[(qbase + t) * DD + d] = (__bf16)aq[i][r];
      k[(qbase + t) * DD + d] = (__bf16)ak[i][r];
    }
    v8bf vp;
#pragma unroll
    for (int r = 0; r < 8; ++r) vp[r] = (__bf16)av[i][r];
    *(v8bf*)(vT + ((size_t)bh * DD + d) * TT + tbase + i * 16 + mloc) = vp;
  }
}

// ---------------------------------------------------------------------------
// Flash attention (causal).  One wave per (bh, 16-query tile).
// S^T = K*Q^T so softmax rows (queries) sit on lanes; O^T = V^T * P^T.
// Writes x2 = x + attn_out (fp32, (B,T,C)).
// ---------------------------------------------------------------------------
__global__ void attn_kernel(const __bf16* __restrict__ q,
                            const __bf16* __restrict__ k,
                            const __bf16* __restrict__ vT,
                            const float* __restrict__ x,
                            float* __restrict__ x2) {
  int wv   = threadIdx.x >> 5;
  int lane = threadIdx.x & 31;
  int ttile = blockIdx.x * 4 + wv;       // 0..T/16-1
  int bh = blockIdx.y;
  int b = bh / HH, hh = bh % HH;
  int tbase = ttile * 16;
  const float scale = 0.03608439182435161f;   // 1/sqrt(C)

  size_t qoff = ((size_t)bh * TT + tbase) * DD;
  v16bf qb0 = load_b(q + qoff, DD);           // d = 0..31
  v16bf qb1 = load_b(q + qoff + 32, DD);      // d = 32..63

  v8f accO[4] = {};
  float mrun = -3.0e38f, lrun = 0.f;
  int n_t  = lane & 15;
  int mloc = (lane >> 4) << 3;
  int tg = tbase + n_t;
  bool lohalf = lane < 16;

  for (int s0 = 0; s0 <= tbase; s0 += 32) {
    v8f st0 = {};
    {
      size_t koff = ((size_t)bh * TT + s0) * DD;
      st0 = wmma_bf16(load_a(k + koff, DD), qb0, st0);
      st0 = wmma_bf16(load_a(k + koff + 32, DD), qb1, st0);
    }
    bool have1 = (s0 + 16) <= (tbase + 15);
    v8f st1 = {};
    if (have1) {
      size_t koff = ((size_t)bh * TT + s0 + 16) * DD;
      st1 = wmma_bf16(load_a(k + koff, DD), qb0, st1);
      st1 = wmma_bf16(load_a(k + koff + 32, DD), qb1, st1);
    }

    float p0[8], p1[8];
    float tmax = -3.0e38f;
#pragma unroll
    for (int r = 0; r < 8; ++r) {
      int sg = s0 + mloc + r;
      float v0 = (sg <= tg) ? st0[r] * scale : -3.0e38f;
      float v1 = (have1 && (sg + 16) <= tg) ? st1[r] * scale : -3.0e38f;
      p0[r] = v0; p1[r] = v1;
      tmax = fmaxf(tmax, fmaxf(v0, v1));
    }
    tmax = fmaxf(tmax, __shfl_xor(tmax, 16, 32));
    float mnew  = fmaxf(mrun, tmax);
    float alpha = __expf(mrun - mnew);
    float psum = 0.f;
#pragma unroll
    for (int r = 0; r < 8; ++r) {
      p0[r] = __expf(p0[r] - mnew);
      p1[r] = __expf(p1[r] - mnew);
      psum += p0[r] + p1[r];
    }
    psum += __shfl_xor(psum, 16, 32);
    lrun = lrun * alpha + psum;
    mrun = mnew;
#pragma unroll
    for (int dt = 0; dt < 4; ++dt)
#pragma unroll
      for (int r = 0; r < 8; ++r) accO[dt][r] *= alpha;

    // Re-layout P^T (two C-layout 16x16 tiles) into the 32x16 B operand:
    // half-wave swap of the m-halves, then lane-local f32->bf16 packing.
    v16bf pb;
#pragma unroll
    for (int e = 0; e < 8; ++e) {
      float o0 = __shfl_xor(p0[e], 16, 32);
      float o1 = __shfl_xor(p1[e], 16, 32);
      pb[e]     = (__bf16)(lohalf ? p0[e] : o1);
      pb[e + 8] = (__bf16)(lohalf ? o0 : p1[e]);
    }
#pragma unroll
    for (int dt = 0; dt < 4; ++dt) {
      v16bf va = load_a(vT + ((size_t)bh * DD + dt * 16) * TT + s0, TT);
      accO[dt] = wmma_bf16(va, pb, accO[dt]);
    }
  }

  float inv = 1.0f / lrun;
  size_t orow = ((size_t)b * TT + tg) * CC;
#pragma unroll
  for (int dt = 0; dt < 4; ++dt)
#pragma unroll
    for (int r = 0; r < 8; ++r) {
      int cidx = hh * DD + dt * 16 + mloc + r;
      x2[orow + cidx] = x[orow + cidx] + accO[dt][r] * inv;
    }
}

// ---------------------------------------------------------------------------
// MLP GEMM 1: hid = relu(h2 @ W_h + b_h).  A (M,C) bf16, Bt (F,C) bf16.
// One wave per 32x64 output tile: 8 WMMA per K-step from 6 operand loads.
// out bf16 (M,F).
// ---------------------------------------------------------------------------
__global__ void mlp1_kernel(const __bf16* __restrict__ h2,
                            const __bf16* __restrict__ whT,
                            const float* __restrict__ bias,
                            __bf16* __restrict__ hid) {
  int wv = threadIdx.x >> 5, lane = threadIdx.x & 31;
  int ng = blockIdx.x * 4 + wv;          // 0..F/64-1  (64-col group)
  int mg = blockIdx.y;                   // 0..M/32-1  (32-row group)
  const __bf16* A  = h2  + (size_t)mg * 32 * CC;
  const __bf16* Bt = whT + (size_t)ng * 64 * CC;
  v8f acc[2][4] = {};
  for (int c0 = 0; c0 < CC; c0 += 32) {
    v16bf a0 = load_a(A + c0, CC);
    v16bf a1 = load_a(A + (size_t)16 * CC + c0, CC);
#pragma unroll
    for (int j = 0; j < 4; ++j) {
      v16bf bt = load_b(Bt + (size_t)j * 16 * CC + c0, CC);
      acc[0][j] = wmma_bf16(a0, bt, acc[0][j]);
      acc[1][j] = wmma_bf16(a1, bt, acc[1][j]);
    }
  }
  int mloc = (lane >> 4) << 3;
  int nl = lane & 15;
#pragma unroll
  for (int j = 0; j < 4; ++j) {
    int n = ng * 64 + j * 16 + nl;
    float bv = bias[n];
#pragma unroll
    for (int i = 0; i < 2; ++i)
#pragma unroll
      for (int r = 0; r < 8; ++r) {
        float v = acc[i][j][r] + bv;
        v = v > 0.f ? v : 0.f;
        hid[(size_t)(mg * 32 + i * 16 + mloc + r) * FF + n] = (__bf16)v;
      }
  }
}

// ---------------------------------------------------------------------------
// MLP GEMM 2 + residual: out = hid @ W_p + b_p + x2.  fp32 output.
// One wave per 32x64 output tile.
// ---------------------------------------------------------------------------
__global__ void mlp2_kernel(const __bf16* __restrict__ hid,
                            const __bf16* __restrict__ wpT,
                            const float* __restrict__ bias,
                            const float* __restrict__ x2,
                            float* __restrict__ out) {
  int wv = threadIdx.x >> 5, lane = threadIdx.x & 31;
  int ng = blockIdx.x * 4 + wv;          // 0..C/64-1
  int mg = blockIdx.y;                   // 0..M/32-1
  const __bf16* A  = hid + (size_t)mg * 32 * FF;
  const __bf16* Bt = wpT + (size_t)ng * 64 * FF;
  v8f acc[2][4] = {};
  for (int c0 = 0; c0 < FF; c0 += 32) {
    v16bf a0 = load_a(A + c0, FF);
    v16bf a1 = load_a(A + (size_t)16 * FF + c0, FF);
#pragma unroll
    for (int j = 0; j < 4; ++j) {
      v16bf bt = load_b(Bt + (size_t)j * 16 * FF + c0, FF);
      acc[0][j] = wmma_bf16(a0, bt, acc[0][j]);
      acc[1][j] = wmma_bf16(a1, bt, acc[1][j]);
    }
  }
  int mloc = (lane >> 4) << 3;
  int nl = lane & 15;
#pragma unroll
  for (int j = 0; j < 4; ++j) {
    int n = ng * 64 + j * 16 + nl;
    float bv = bias[n];
#pragma unroll
    for (int i = 0; i < 2; ++i)
#pragma unroll
      for (int r = 0; r < 8; ++r) {
        size_t idx = (size_t)(mg * 32 + i * 16 + mloc + r) * CC + n;
        out[idx] = acc[i][j][r] + bv + x2[idx];
      }
  }
}

// ---------------------------------------------------------------------------
extern "C" void kernel_launch(void* const* d_in, const int* in_sizes, int n_in,
                              void* d_out, int out_size, void* d_ws,
                              size_t ws_size, hipStream_t stream) {
  const float* x        = (const float*)d_in[0];
  const float* ln1_g    = (const float*)d_in[1];
  const float* ln1_b    = (const float*)d_in[2];
  const float* wq       = (const float*)d_in[3];
  const float* wk       = (const float*)d_in[4];
  const float* wvp      = (const float*)d_in[5];
  const float* ln2_g    = (const float*)d_in[6];
  const float* ln2_b    = (const float*)d_in[7];
  const float* w_hidden = (const float*)d_in[8];
  const float* b_hidden = (const float*)d_in[9];
  const float* w_proj   = (const float*)d_in[10];
  const float* b_proj   = (const float*)d_in[11];

  char* ws = (char*)d_ws;
  const size_t SZ_WQT = (size_t)HH * DD * CC * 2;   // 1.125 MB each
  const size_t SZ_WHT = (size_t)CC * FF * 2;        // 4.5 MB each
  const size_t SZ_ACT = (size_t)MM * CC * 2;        // 6 MB bf16 activation
  const size_t SZ_QKV = (size_t)BB * HH * TT * DD * 2;
  size_t off = 0;
  __bf16* wqT = (__bf16*)(ws + off); off += SZ_WQT;
  __bf16* wkT = (__bf16*)(ws + off); off += SZ_WQT;
  __bf16* wvT = (__bf16*)(ws + off); off += SZ_WQT;
  __bf16* whT = (__bf16*)(ws + off); off += SZ_WHT;
  __bf16* wpT = (__bf16*)(ws + off); off += SZ_WHT;
  size_t off_h1 = off;
  __bf16* h1  = (__bf16*)(ws + off); off += SZ_ACT;
  __bf16* qb  = (__bf16*)(ws + off); off += SZ_QKV;
  __bf16* kb  = (__bf16*)(ws + off); off += SZ_QKV;
  __bf16* vT  = (__bf16*)(ws + off); off += SZ_QKV;
  float*  x2  = (float*)(ws + off);  off += (size_t)MM * CC * 4;
  __bf16* h2  = (__bf16*)(ws + off); off += SZ_ACT;
  __bf16* hid = (__bf16*)(ws + off_h1);  // aliases dead h1/q/k/vT (25.2 MB)

  // Stage weights as bf16, N-major.
  transpose_cvt_kernel<<<dim3((CC * DD + 255) / 256, HH), 256, 0, stream>>>(wq, wqT, CC, DD);
  transpose_cvt_kernel<<<dim3((CC * DD + 255) / 256, HH), 256, 0, stream>>>(wk, wkT, CC, DD);
  transpose_cvt_kernel<<<dim3((CC * DD + 255) / 256, HH), 256, 0, stream>>>(wvp, wvT, CC, DD);
  transpose_cvt_kernel<<<dim3((CC * FF + 255) / 256, 1), 256, 0, stream>>>(w_hidden, whT, CC, FF);
  transpose_cvt_kernel<<<dim3((FF * CC + 255) / 256, 1), 256, 0, stream>>>(w_proj, wpT, FF, CC);

  // LN1 -> QKV -> flash attention (+residual) -> LN2 -> MLP (+residual).
  ln_kernel<<<MM, 256, 0, stream>>>(x, ln1_g, ln1_b, h1);
  qkv_kernel<<<dim3(TT / 32, BB * HH), 128, 0, stream>>>(h1, wqT, wkT, wvT, qb, kb, vT);
  attn_kernel<<<dim3(TT / 64, BB * HH), 128, 0, stream>>>(qb, kb, vT, x, x2);
  ln_kernel<<<MM, 256, 0, stream>>>(x2, ln2_g, ln2_b, h2);
  mlp1_kernel<<<dim3(FF / 256, MM / 32), 128, 0, stream>>>(h2, whT, b_hidden, hid);
  mlp2_kernel<<<dim3(CC / 256, MM / 32), 128, 0, stream>>>(hid, wpT, b_proj, x2, (float*)d_out);
}
